// AAModel_29506425324139
// MI455X (gfx1250) — matrix operand order repaired and checked by reading, not compiled, not run on hardware
//
#include <hip/hip_runtime.h>

typedef __attribute__((ext_vector_type(16))) __bf16 v16bf;
typedef __attribute__((ext_vector_type(8)))  float  v8f;

#define NSD 48
#define NVD 10
#define KDIM 144      // 3*NS
#define KPAD 160      // padded K for 5 x K=32 WMMA steps
#define KT_STEPS 5
#define NT1 9         // 144/16 hidden tiles
#define NT2A 174      // 2784/16
#define NT2B 217      // ceil(3464/16)
#define W1COLS 2784
#define W2COLS 3464
#define EPB 32        // edges (M rows) per block: 2 x 16-row WMMA tiles share B
#define FRAG_HALFS (KT_STEPS * 32 * 16)   // bf16 elems in one 16xKPAD A buffer

__device__ __forceinline__ unsigned short f2bf_bits(float f) {
  unsigned u = __builtin_bit_cast(unsigned, f);
  unsigned r = u + 0x7FFFu + ((u >> 16) & 1u);   // round-to-nearest-even
  return (unsigned short)(r >> 16);
}
__device__ __forceinline__ __bf16 f2bf(float f) {
  unsigned short s = f2bf_bits(f);
  return __builtin_bit_cast(__bf16, s);
}

// A-fragment slot for element (m, k) of a 16 x KPAD matrix.
// Loader reads lane l -> row m=l&15, klo=(l>>4)*8; elem j -> k = kt*32 +
// (j<8 ? klo+j : 16+klo+j-8). Inverse (kk = k%32):
//   lane = m + 16*((kk>>3)&1),  j = (kk&7) + 8*(kk>>4)
__device__ __forceinline__ int afrag_slot(int m, int k) {
  int kt = k >> 5, kk = k & 31;
  int lane = m + (((kk >> 3) & 1) << 4);
  int j = (kk & 7) + ((kk >> 4) << 3);
  return (kt * 32 + lane) * 16 + j;
}

// ---- pre-pack weights [K,N] f32 (row-major) into bf16 B-fragment groups ----
__global__ void k_pack(const float* __restrict__ W, unsigned short* __restrict__ frag,
                       int K, int Ncols, int NT) {
  int gid = blockIdx.x * blockDim.x + threadIdx.x;
  int total = KT_STEPS * NT * 32;
  if (gid >= total) return;
  int lane = gid & 31;
  int t  = (gid >> 5) % NT;
  int kt = (gid >> 5) / NT;
  int n = t * 16 + (lane & 15);
  int kbase = kt * 32 + ((lane >> 4) * 8);
  unsigned short* out = frag + (size_t)gid * 16;
  #pragma unroll
  for (int j = 0; j < 16; ++j) {
    int k = kbase + (j < 8 ? j : 8 + j);
    float v = (k < K && n < Ncols) ? W[(size_t)k * Ncols + n] : 0.0f;
    out[j] = f2bf_bits(v);
  }
}

// ---- per-edge: sh1, gaussian smearing, edge MLP 64->48->48, degree count ----
__global__ void __launch_bounds__(256) k_edge_pre(
    const float* __restrict__ pos, const float* __restrict__ sig,
    const float* __restrict__ eW1, const float* __restrict__ eb1,
    const float* __restrict__ eW2, const float* __restrict__ eb2,
    const int* __restrict__ ei,
    float* __restrict__ sh1, float* __restrict__ edge_emb, float* deg, int E)
{
  __shared__ float sW1[64 * 48];
  __shared__ float sW2[48 * 48];
  __shared__ float sb1[48];
  __shared__ float sb2[48];
  int tid = threadIdx.x;
  for (int i = tid; i < 64 * 48; i += 256) sW1[i] = eW1[i];
  for (int i = tid; i < 48 * 48; i += 256) sW2[i] = eW2[i];
  if (tid < 48) { sb1[tid] = eb1[tid]; sb2[tid] = eb2[tid]; }
  __syncthreads();
  int e = blockIdx.x * 256 + tid;
  if (e >= E) return;
  int s = ei[e], dn = ei[E + e];
  float vx = pos[(size_t)dn * 3 + 0] - pos[(size_t)s * 3 + 0];
  float vy = pos[(size_t)dn * 3 + 1] - pos[(size_t)s * 3 + 1];
  float vz = pos[(size_t)dn * 3 + 2] - pos[(size_t)s * 3 + 2];
  float d = sqrtf(vx * vx + vy * vy + vz * vz);
  float ish = 1.7320508075688772f / (d + 1e-8f);   // sqrt(3)/(d+eps)
  sh1[(size_t)e * 3 + 0] = vx * ish;
  sh1[(size_t)e * 3 + 1] = vy * ish;
  sh1[(size_t)e * 3 + 2] = vz * ish;

  float x[64];
  #pragma unroll
  for (int i = 0; i < 32; ++i) x[i] = sig[(size_t)s * 32 + i];
  const float delta = 30.0f / 31.0f;
  const float coeff = -0.5f / (delta * delta);
  #pragma unroll
  for (int k = 0; k < 32; ++k) {
    float t = d - delta * (float)k;
    x[32 + k] = __expf(coeff * t * t);
  }
  float h[48];
  #pragma unroll
  for (int o = 0; o < 48; ++o) h[o] = sb1[o];
  for (int i = 0; i < 64; ++i) {
    float xi = x[i];
    #pragma unroll
    for (int o = 0; o < 48; ++o) h[o] = fmaf(xi, sW1[i * 48 + o], h[o]);
  }
  #pragma unroll
  for (int o = 0; o < 48; ++o) h[o] = fmaxf(h[o], 0.0f);
  for (int o2 = 0; o2 < 48; ++o2) {
    float acc = sb2[o2];
    for (int o = 0; o < 48; ++o) acc = fmaf(h[o], sW2[o * 48 + o2], acc);
    edge_emb[(size_t)e * 48 + o2] = acc;
  }
  atomicAdd(&deg[dn], 1.0f);
}

// ---- conv layer 1: fused weight-gen GEMM (WMMA) + TP contraction + scatter ----
__global__ void __launch_bounds__(256) k_conv1(
    const float* __restrict__ node_s, const int* __restrict__ ei,
    const float* __restrict__ sh1g, const float* __restrict__ edge_emb,
    const unsigned short* __restrict__ fragA, const float* __restrict__ fb1,
    const unsigned short* __restrict__ fragB, const float* __restrict__ fb2,
    float* acc_s, float* acc_v, int E)
{
  __shared__ __attribute__((aligned(32))) __bf16 eaf[2][FRAG_HALFS];
  __shared__ __attribute__((aligned(32))) __bf16 hmf[2][FRAG_HALFS];
  __shared__ float xs[EPB * NSD];
  __shared__ float ms[EPB * NSD];
  __shared__ float sv[EPB * NVD];
  __shared__ float shl[EPB * 3];
  __shared__ int srcl[EPB], dstl[EPB];
  int tid = threadIdx.x, lane = tid & 31, wave = tid >> 5;
  int e0 = blockIdx.x * EPB;

  if (tid < EPB) {
    int e = e0 + tid;
    srcl[tid] = (e < E) ? ei[e] : 0;
    dstl[tid] = (e < E) ? ei[E + e] : 0;
  }
  __syncthreads();
  for (int idx = tid; idx < EPB * KPAD; idx += 256) {
    int m = idx / KPAD, c = idx - m * KPAD;
    int e = e0 + m;
    float v = 0.f;
    if (e < E) {
      if (c < 48)       v = edge_emb[(size_t)e * 48 + c];
      else if (c < 96)  v = node_s[(size_t)srcl[m] * 48 + (c - 48)];
      else if (c < 144) v = node_s[(size_t)dstl[m] * 48 + (c - 96)];
    }
    eaf[m >> 4][afrag_slot(m & 15, c)] = f2bf(v);
    if (c >= 48 && c < 96) xs[m * NSD + (c - 48)] = v;
  }
  for (int idx = tid; idx < 2 * FRAG_HALFS; idx += 256) hmf[0][idx] = f2bf(0.f);
  for (int idx = tid; idx < EPB * NSD; idx += 256) ms[idx] = 0.f;
  for (int idx = tid; idx < EPB * NVD; idx += 256) sv[idx] = 0.f;
  for (int idx = tid; idx < EPB * 3; idx += 256) {
    int m = idx / 3, e = e0 + m;
    shl[idx] = (e < E) ? sh1g[(size_t)e * 3 + (idx - m * 3)] : 0.f;
  }
  __syncthreads();

  // stage 1: h = relu(ea @ W1 + b1), [EPB,144] -> hmf (bf16 frag order)
  {
    v16bf a0[KT_STEPS], a1[KT_STEPS];
    #pragma unroll
    for (int kt = 0; kt < KT_STEPS; ++kt) {
      a0[kt] = *reinterpret_cast<const v16bf*>(&eaf[0][(kt * 32 + lane) * 16]);
      a1[kt] = *reinterpret_cast<const v16bf*>(&eaf[1][(kt * 32 + lane) * 16]);
    }
    for (int t = wave; t < NT1; t += 8) {
      v8f c0 = {}, c1 = {};
      #pragma unroll
      for (int kt = 0; kt < KT_STEPS; ++kt) {
        v16bf b = *reinterpret_cast<const v16bf*>(fragA + ((size_t)(kt * NT1 + t) * 32 + lane) * 16);
        c0 = __builtin_amdgcn_wmma_f32_16x16x32_bf16(false, a0[kt], false, b, (short)0, c0, false, false);
        c1 = __builtin_amdgcn_wmma_f32_16x16x32_bf16(false, a1[kt], false, b, (short)0, c1, false, false);
      }
      int n = lane & 15, mbase = (lane >> 4) * 8;
      int col = t * 16 + n;
      float bias = fb1[col];
      #pragma unroll
      for (int r = 0; r < 8; ++r) {
        int slot = afrag_slot(mbase + r, col);
        hmf[0][slot] = f2bf(fmaxf(c0[r] + bias, 0.f));
        hmf[1][slot] = f2bf(fmaxf(c1[r] + bias, 0.f));
      }
    }
  }
  __syncthreads();

  // stage 2: w = h @ W2 + b2 (never materialized): contract tile-by-tile
  {
    v16bf a0[KT_STEPS], a1[KT_STEPS];
    #pragma unroll
    for (int kt = 0; kt < KT_STEPS; ++kt) {
      a0[kt] = *reinterpret_cast<const v16bf*>(&hmf[0][(kt * 32 + lane) * 16]);
      a1[kt] = *reinterpret_cast<const v16bf*>(&hmf[1][(kt * 32 + lane) * 16]);
    }
    for (int t = wave; t < NT2A; t += 8) {
      v8f c0 = {}, c1 = {};
      #pragma unroll
      for (int kt = 0; kt < KT_STEPS; ++kt) {
        v16bf b = *reinterpret_cast<const v16bf*>(fragB + ((size_t)(kt * NT2A + t) * 32 + lane) * 16);
        c0 = __builtin_amdgcn_wmma_f32_16x16x32_bf16(false, a0[kt], false, b, (short)0, c0, false, false);
        c1 = __builtin_amdgcn_wmma_f32_16x16x32_bf16(false, a1[kt], false, b, (short)0, c1, false, false);
      }
      int n = lane & 15, mbase = (lane >> 4) * 8;
      int col = t * 16 + n;
      float bias = fb2[col];
      int i, o;
      if (col < 2304) { i = col / 48; o = col - i * 48; }
      else { int q = col - 2304; i = q / 10; o = q - i * 10; }
      #pragma unroll
      for (int h = 0; h < 2; ++h) {
        #pragma unroll
        for (int r = 0; r < 8; ++r) {
          float val = ((h == 0) ? c0[r] : c1[r]) + bias;
          int mrow = h * 16 + mbase + r;
          if (col < 2304) {                       // w00 [48,48]: m_s
            atomicAdd(&ms[mrow * NSD + o], val * xs[mrow * NSD + i]);
          } else {                                // w01 [48,10]: m_v scalar part
            atomicAdd(&sv[mrow * NVD + o], val * xs[mrow * NSD + i]);
          }
        }
      }
    }
  }
  __syncthreads();

  const float invs = 0.14433756729740643f;    // 1/sqrt(48)
  for (int idx = tid; idx < EPB * NSD; idx += 256) {
    int m = idx / NSD, e = e0 + m;
    if (e < E) atomicAdd(&acc_s[(size_t)dstl[m] * NSD + (idx - m * NSD)], ms[idx] * invs);
  }
  for (int idx = tid; idx < EPB * 30; idx += 256) {
    int m = idx / 30, r = idx - m * 30, o = r / 3, d = r - o * 3;
    int e = e0 + m;
    if (e < E) atomicAdd(&acc_v[(size_t)dstl[m] * 30 + r], sv[m * NVD + o] * shl[m * 3 + d] * invs);
  }
}

__global__ void k_fin1(const float* __restrict__ node_s, const float* __restrict__ acc_s,
                       const float* __restrict__ acc_v, const float* __restrict__ deg,
                       float* node_s1, float* node_v1, int N) {
  int idx = blockIdx.x * blockDim.x + threadIdx.x;
  if (idx >= N * 48) return;
  int n = idx / 48, j = idx - n * 48;
  float c = fmaxf(deg[n], 1.0f);
  node_s1[idx] = node_s[idx] + acc_s[idx] / c;
  if (j < 30) node_v1[(size_t)n * 30 + j] = acc_v[(size_t)n * 30 + j] / c;
}

// ---- conv layer 2 ----
__global__ void __launch_bounds__(256) k_conv2(
    const float* __restrict__ node_s1, const float* __restrict__ node_v1,
    const int* __restrict__ ei, const float* __restrict__ sh1g,
    const float* __restrict__ edge_emb,
    const unsigned short* __restrict__ fragA, const float* __restrict__ fb1,
    const unsigned short* __restrict__ fragB, const float* __restrict__ fb2,
    float* acc_s2, float* acc_v2, float* acc_p, int E)
{
  __shared__ __attribute__((aligned(32))) __bf16 eaf[2][FRAG_HALFS];
  __shared__ __attribute__((aligned(32))) __bf16 hmf[2][FRAG_HALFS];
  __shared__ float xs[EPB * NSD];
  __shared__ float ms[EPB * NSD];
  __shared__ float sv[EPB * NVD];
  __shared__ float mv[EPB * NVD * 3];
  __shared__ float mp[EPB * NVD * 3];
  __shared__ float xv[EPB * NVD * 3];
  __shared__ float xdot[EPB * NVD];
  __shared__ float xcr[EPB * NVD * 3];
  __shared__ float shl[EPB * 3];
  __shared__ int srcl[EPB], dstl[EPB];
  int tid = threadIdx.x, lane = tid & 31, wave = tid >> 5;
  int e0 = blockIdx.x * EPB;

  if (tid < EPB) {
    int e = e0 + tid;
    srcl[tid] = (e < E) ? ei[e] : 0;
    dstl[tid] = (e < E) ? ei[E + e] : 0;
  }
  __syncthreads();
  for (int idx = tid; idx < EPB * KPAD; idx += 256) {
    int m = idx / KPAD, c = idx - m * KPAD;
    int e = e0 + m;
    float v = 0.f;
    if (e < E) {
      if (c < 48)       v = edge_emb[(size_t)e * 48 + c];
      else if (c < 96)  v = node_s1[(size_t)srcl[m] * 48 + (c - 48)];
      else if (c < 144) v = node_s1[(size_t)dstl[m] * 48 + (c - 96)];
    }
    eaf[m >> 4][afrag_slot(m & 15, c)] = f2bf(v);
    if (c >= 48 && c < 96) xs[m * NSD + (c - 48)] = v;
  }
  for (int idx = tid; idx < 2 * FRAG_HALFS; idx += 256) hmf[0][idx] = f2bf(0.f);
  for (int idx = tid; idx < EPB * 30; idx += 256) {
    int m = idx / 30, r = idx - m * 30;
    int e = e0 + m;
    xv[idx] = (e < E) ? node_v1[(size_t)srcl[m] * 30 + r] : 0.f;
    mv[idx] = 0.f;
    mp[idx] = 0.f;
  }
  for (int idx = tid; idx < EPB * NSD; idx += 256) ms[idx] = 0.f;
  for (int idx = tid; idx < EPB * NVD; idx += 256) sv[idx] = 0.f;
  for (int idx = tid; idx < EPB * 3; idx += 256) {
    int m = idx / 3, e = e0 + m;
    shl[idx] = (e < E) ? sh1g[(size_t)e * 3 + (idx - m * 3)] : 0.f;
  }
  __syncthreads();
  for (int idx = tid; idx < EPB * NVD; idx += 256) {   // xv.sh1 and xv x sh1
    int m = idx / NVD, i = idx - m * NVD;
    const float* a = &xv[(m * NVD + i) * 3];
    float sx = shl[m * 3 + 0], sy = shl[m * 3 + 1], sz = shl[m * 3 + 2];
    xdot[idx] = a[0] * sx + a[1] * sy + a[2] * sz;
    float* cr = &xcr[(m * NVD + i) * 3];
    cr[0] = a[1] * sz - a[2] * sy;
    cr[1] = a[2] * sx - a[0] * sz;
    cr[2] = a[0] * sy - a[1] * sx;
  }

  // stage 1
  {
    v16bf a0[KT_STEPS], a1[KT_STEPS];
    #pragma unroll
    for (int kt = 0; kt < KT_STEPS; ++kt) {
      a0[kt] = *reinterpret_cast<const v16bf*>(&eaf[0][(kt * 32 + lane) * 16]);
      a1[kt] = *reinterpret_cast<const v16bf*>(&eaf[1][(kt * 32 + lane) * 16]);
    }
    for (int t = wave; t < NT1; t += 8) {
      v8f c0 = {}, c1 = {};
      #pragma unroll
      for (int kt = 0; kt < KT_STEPS; ++kt) {
        v16bf b = *reinterpret_cast<const v16bf*>(fragA + ((size_t)(kt * NT1 + t) * 32 + lane) * 16);
        c0 = __builtin_amdgcn_wmma_f32_16x16x32_bf16(false, a0[kt], false, b, (short)0, c0, false, false);
        c1 = __builtin_amdgcn_wmma_f32_16x16x32_bf16(false, a1[kt], false, b, (short)0, c1, false, false);
      }
      int n = lane & 15, mbase = (lane >> 4) * 8;
      int col = t * 16 + n;
      float bias = fb1[col];
      #pragma unroll
      for (int r = 0; r < 8; ++r) {
        int slot = afrag_slot(mbase + r, col);
        hmf[0][slot] = f2bf(fmaxf(c0[r] + bias, 0.f));
        hmf[1][slot] = f2bf(fmaxf(c1[r] + bias, 0.f));
      }
    }
  }
  __syncthreads();

  // stage 2: 3464 generated weights, five TP regions
  const float nA  = 0.10206207261596575f;   // 1/sqrt(96)
  const float nB  = 0.22360679774997896f;   // 1/sqrt(20)
  const float nBs = 0.12909944487358055f;   // nB/sqrt(3)
  {
    v16bf a0[KT_STEPS], a1[KT_STEPS];
    #pragma unroll
    for (int kt = 0; kt < KT_STEPS; ++kt) {
      a0[kt] = *reinterpret_cast<const v16bf*>(&hmf[0][(kt * 32 + lane) * 16]);
      a1[kt] = *reinterpret_cast<const v16bf*>(&hmf[1][(kt * 32 + lane) * 16]);
    }
    for (int t = wave; t < NT2B; t += 8) {
      v8f c0 = {}, c1 = {};
      #pragma unroll
      for (int kt = 0; kt < KT_STEPS; ++kt) {
        v16bf b = *reinterpret_cast<const v16bf*>(fragB + ((size_t)(kt * NT2B + t) * 32 + lane) * 16);
        c0 = __builtin_amdgcn_wmma_f32_16x16x32_bf16(false, a0[kt], false, b, (short)0, c0, false, false);
        c1 = __builtin_amdgcn_wmma_f32_16x16x32_bf16(false, a1[kt], false, b, (short)0, c1, false, false);
      }
      int n = lane & 15, mbase = (lane >> 4) * 8;
      int col = t * 16 + n;
      if (col < W2COLS) {
        float bias = fb2[col];
        #pragma unroll
        for (int h = 0; h < 2; ++h) {
          #pragma unroll
          for (int r = 0; r < 8; ++r) {
            float val = ((h == 0) ? c0[r] : c1[r]) + bias;
            int mrow = h * 16 + mbase + r;
            if (col < 2304) {                       // w00 [48,48] (x) xs -> 0e
              int i = col / 48, o = col - i * 48;
              atomicAdd(&ms[mrow * NSD + o], val * xs[mrow * NSD + i] * nA);
            } else if (col < 2784) {                // w01 [48,10] (x) xs -> 1o scalar
              int q = col - 2304, i = q / 10, o = q - i * 10;
              atomicAdd(&sv[mrow * NVD + o], val * xs[mrow * NSD + i] * nA);
            } else if (col < 2884) {                // w10 [10,10] (x) xv -> 1o
              int q = col - 2784, i = q / 10, o = q - i * 10;
              const float* xb = &xv[(mrow * NVD + i) * 3];
              float* mb = &mv[(mrow * NVD + o) * 3];
              atomicAdd(mb + 0, val * xb[0] * nB);
              atomicAdd(mb + 1, val * xb[1] * nB);
              atomicAdd(mb + 2, val * xb[2] * nB);
            } else if (col < 3364) {                // w11s [10,48] (x) (xv.sh) -> 0e
              int q = col - 2884, i = q / 48, o = q - i * 48;
              atomicAdd(&ms[mrow * NSD + o], val * xdot[mrow * NVD + i] * nBs);
            } else {                                // w11p [10,10] (x) (xv x sh) -> 1e
              int q = col - 3364, i = q / 10, o = q - i * 10;
              const float* cb = &xcr[(mrow * NVD + i) * 3];
              float* mb = &mp[(mrow * NVD + o) * 3];
              atomicAdd(mb + 0, val * cb[0] * nB);
              atomicAdd(mb + 1, val * cb[1] * nB);
              atomicAdd(mb + 2, val * cb[2] * nB);
            }
          }
        }
      }
    }
  }
  __syncthreads();

  for (int idx = tid; idx < EPB * NSD; idx += 256) {
    int m = idx / NSD, e = e0 + m;
    if (e < E) atomicAdd(&acc_s2[(size_t)dstl[m] * NSD + (idx - m * NSD)], ms[idx]);
  }
  for (int idx = tid; idx < EPB * 30; idx += 256) {
    int m = idx / 30, r = idx - m * 30, o = r / 3, d = r - o * 3;
    int e = e0 + m;
    if (e < E) {
      int dn = dstl[m];
      atomicAdd(&acc_v2[(size_t)dn * 30 + r], sv[m * NVD + o] * shl[m * 3 + d] + mv[idx]);
      atomicAdd(&acc_p [(size_t)dn * 30 + r], mp[idx]);
    }
  }
}

__global__ void k_fin2(const float* __restrict__ node_s1, const float* __restrict__ node_v1,
                       const float* __restrict__ acc_s2, const float* __restrict__ acc_v2,
                       const float* __restrict__ acc_p, const float* __restrict__ deg,
                       float* out, int N) {
  int idx = blockIdx.x * blockDim.x + threadIdx.x;
  if (idx >= N * 108) return;
  int n = idx / 108, j = idx - n * 108;
  float c = fmaxf(deg[n], 1.0f);
  float v;
  if (j < 48)       v = node_s1[(size_t)n * 48 + j] + acc_s2[(size_t)n * 48 + j] / c;
  else if (j < 78)  v = node_v1[(size_t)n * 30 + (j - 48)] + acc_v2[(size_t)n * 30 + (j - 48)] / c;
  else              v = acc_p[(size_t)n * 30 + (j - 78)] / c;
  out[idx] = v;
}

extern "C" void kernel_launch(void* const* d_in, const int* in_sizes, int n_in,
                              void* d_out, int out_size, void* d_ws, size_t ws_size,
                              hipStream_t stream) {
  const float* pos  = (const float*)d_in[0];
  const float* sig  = (const float*)d_in[1];
  const float* ns0  = (const float*)d_in[2];
  const float* eW1  = (const float*)d_in[3];
  const float* eb1  = (const float*)d_in[4];
  const float* eW2  = (const float*)d_in[5];
  const float* eb2  = (const float*)d_in[6];
  const float* f1W1 = (const float*)d_in[7];
  const float* f1b1 = (const float*)d_in[8];
  const float* f1W2 = (const float*)d_in[9];
  const float* f1b2 = (const float*)d_in[10];
  const float* f2W1 = (const float*)d_in[11];
  const float* f2b1 = (const float*)d_in[12];
  const float* f2W2 = (const float*)d_in[13];
  const float* f2b2 = (const float*)d_in[14];
  const int*   ei   = (const int*)d_in[15];
  float* out = (float*)d_out;
  int N = in_sizes[0] / 3;
  int E = in_sizes[15] / 2;

  char* base = (char*)d_ws;
  size_t off = 0;
  auto take = [&](size_t bytes) {
    char* r = base + off;
    off = (off + bytes + 255) & ~(size_t)255;
    return r;
  };
  float* sh1      = (float*)take((size_t)E * 3 * 4);
  float* edge_emb = (float*)take((size_t)E * 48 * 4);
  float* deg      = (float*)take((size_t)N * 4);
  float* node_s1  = (float*)take((size_t)N * 48 * 4);
  float* node_v1  = (float*)take((size_t)N * 30 * 4);
  float* acc_s    = (float*)take((size_t)N * 48 * 4);   // reused for layer 2
  float* acc_v    = (float*)take((size_t)N * 30 * 4);   // reused for layer 2
  float* acc_p    = (float*)take((size_t)N * 30 * 4);
  unsigned short* frag11 = (unsigned short*)take((size_t)KT_STEPS * NT1  * 32 * 16 * 2);
  unsigned short* frag12 = (unsigned short*)take((size_t)KT_STEPS * NT2A * 32 * 16 * 2);
  unsigned short* frag21 = (unsigned short*)take((size_t)KT_STEPS * NT1  * 32 * 16 * 2);
  unsigned short* frag22 = (unsigned short*)take((size_t)KT_STEPS * NT2B * 32 * 16 * 2);
  (void)ws_size; (void)n_in; (void)out_size;

  hipMemsetAsync(deg,   0, (size_t)N * 4, stream);
  hipMemsetAsync(acc_s, 0, (size_t)N * 48 * 4, stream);
  hipMemsetAsync(acc_v, 0, (size_t)N * 30 * 4, stream);

  auto pg = [](int nt) { return (KT_STEPS * nt * 32 + 255) / 256; };
  k_pack<<<pg(NT1),  256, 0, stream>>>(f1W1, frag11, KDIM, KDIM,   NT1);
  k_pack<<<pg(NT2A), 256, 0, stream>>>(f1W2, frag12, KDIM, W1COLS, NT2A);
  k_pack<<<pg(NT1),  256, 0, stream>>>(f2W1, frag21, KDIM, KDIM,   NT1);
  k_pack<<<pg(NT2B), 256, 0, stream>>>(f2W2, frag22, KDIM, W2COLS, NT2B);

  k_edge_pre<<<(E + 255) / 256, 256, 0, stream>>>(pos, sig, eW1, eb1, eW2, eb2, ei,
                                                  sh1, edge_emb, deg, E);
  k_conv1<<<(E + EPB - 1) / EPB, 256, 0, stream>>>(ns0, ei, sh1, edge_emb,
                                                   frag11, f1b1, frag12, f1b2,
                                                   acc_s, acc_v, E);
  k_fin1<<<(N * 48 + 255) / 256, 256, 0, stream>>>(ns0, acc_s, acc_v, deg,
                                                   node_s1, node_v1, N);

  hipMemsetAsync(acc_s, 0, (size_t)N * 48 * 4, stream);
  hipMemsetAsync(acc_v, 0, (size_t)N * 30 * 4, stream);
  hipMemsetAsync(acc_p, 0, (size_t)N * 30 * 4, stream);

  k_conv2<<<(E + EPB - 1) / EPB, 256, 0, stream>>>(node_s1, node_v1, ei, sh1, edge_emb,
                                                   frag21, f2b1, frag22, f2b2,
                                                   acc_s, acc_v, acc_p, E);
  k_fin2<<<(N * 108 + 255) / 256, 256, 0, stream>>>(node_s1, node_v1, acc_s, acc_v,
                                                    acc_p, deg, out, N);
}
